// AtomAttentionEncoder_34995393528038
// MI455X (gfx1250) — matrix operand order repaired and checked by reading, not compiled, not run on hardware
//
#include <hip/hip_runtime.h>
#include <math.h>

#define NATOM 1536
#define TTOK  48
#define CA    128
#define CP    16
#define CT    384
#define NB    3
#define NH    4
#define DH    32
#define FDIM  1268
#define JSPLIT 4
#define NTILE (NATOM / 16)   // 96

typedef __bf16 bf16_t;
typedef __attribute__((ext_vector_type(16))) __bf16 v16bf;
typedef __attribute__((ext_vector_type(8)))  float  v8f;

__device__ inline v8f wmma_bf16(v16bf a, v16bf b, v8f c) {
  // D = A(16x32 bf16) x B(32x16 bf16) + C(16x16 f32)
  return __builtin_amdgcn_wmma_f32_16x16x32_bf16(false, a, false, b, (short)0, c, false, false);
}

__device__ inline v8f vzero8() {
  v8f z;
#pragma unroll
  for (int r = 0; r < 8; ++r) z[r] = 0.0f;
  return z;
}

__device__ inline float block_reduce_sum128(float v, float* red) {
  const int t = threadIdx.x;
  red[t] = v;
  __syncthreads();
#pragma unroll
  for (int s = 64; s > 0; s >>= 1) {
    if (t < s) red[t] += red[t + s];
    __syncthreads();
  }
  float r = red[0];
  __syncthreads();
  return r;
}

// ---------------------------------------------------------------------------
// Stage 1: c0, q0 = c0 + r_t@Wnp^T, c = c0 + LN(s_trunk[n%T])@Wts^T,
//          cu1 = c@Wpu1^T, cu2 = c@Wpu2^T.   grid=N, block=128.
// ---------------------------------------------------------------------------
__global__ void k_stage1(const float* __restrict__ ref_pos, const float* __restrict__ r_t,
                         const float* __restrict__ s_trunk, const float* __restrict__ W_feat,
                         const float* __restrict__ ln_g, const float* __restrict__ ln_b,
                         const float* __restrict__ Wts, const float* __restrict__ Wnp,
                         const float* __restrict__ Wpu1, const float* __restrict__ Wpu2,
                         float* __restrict__ qo, float* __restrict__ co,
                         float* __restrict__ cu1, float* __restrict__ cu2) {
  __shared__ float red[128];
  __shared__ float sln[CT];
  __shared__ float sc[CA];
  const int n = blockIdx.x, t = threadIdx.x;

  const float* srow = s_trunk + (n % TTOK) * CT;
  const float x0 = srow[t], x1 = srow[t + 128], x2 = srow[t + 256];
  const float mean = block_reduce_sum128(x0 + x1 + x2, red) * (1.0f / CT);
  const float d0 = x0 - mean, d1 = x1 - mean, d2 = x2 - mean;
  const float var = block_reduce_sum128(d0 * d0 + d1 * d1 + d2 * d2, red) * (1.0f / CT);
  const float rinv = rsqrtf(var + 1e-5f);
  sln[t]       = d0 * rinv * ln_g[t]       + ln_b[t];
  sln[t + 128] = d1 * rinv * ln_g[t + 128] + ln_b[t + 128];
  sln[t + 256] = d2 * rinv * ln_g[t + 256] + ln_b[t + 256];
  __syncthreads();

  const float rp0 = ref_pos[n * 3], rp1 = ref_pos[n * 3 + 1], rp2 = ref_pos[n * 3 + 2];
  const float rt0 = r_t[n * 3], rt1 = r_t[n * 3 + 1], rt2 = r_t[n * 3 + 2];
  const float* wf = W_feat + (size_t)t * FDIM;
  const float c0 = wf[0] * rp0 + wf[1] * rp1 + wf[2] * rp2 + wf[3];
  const float qv = c0 + Wnp[t * 3] * rt0 + Wnp[t * 3 + 1] * rt1 + Wnp[t * 3 + 2] * rt2;

  float acc = 0.0f;
  for (int k = 0; k < CT; ++k) acc += Wts[t * CT + k] * sln[k];
  const float cv = c0 + acc;

  qo[n * CA + t] = qv;
  co[n * CA + t] = cv;
  sc[t] = cv;
  __syncthreads();

  if (t < 2 * CP) {
    const int cp = t & 15;
    const float* wp = (t < CP) ? Wpu1 : Wpu2;
    float a = 0.0f;
    for (int k = 0; k < CA; ++k) a += wp[cp * CA + k] * sc[k];
    if (t < CP) cu1[n * CP + cp] = a; else cu2[n * CP + cp] = a;
  }
}

// ---------------------------------------------------------------------------
// Pair tensor: p_init -> 4-layer MLP fused, one WMMA tile per wave.
// tile = (i, j0..j0+15).  grid = N*(N/16)/4 blocks, 128 threads (4 waves).
// ---------------------------------------------------------------------------
__global__ void k_pair(const float* __restrict__ ref_pos,
                       const float* __restrict__ Wd1, const float* __restrict__ Wd2,
                       const float* __restrict__ Wm1, const float* __restrict__ Wm2,
                       const float* __restrict__ Wm3, const float* __restrict__ Wm4,
                       const float* __restrict__ cu1, const float* __restrict__ cu2,
                       float* __restrict__ p_out) {
  __shared__ float sT[4][16][16];
  const int lane = threadIdx.x & 31;
  const int wid = threadIdx.x >> 5;
  const int tile = blockIdx.x * 4 + wid;
  const int i = tile / NTILE;
  const int j0 = (tile % NTILE) * 16;
  const int half = lane >> 4;
  const int mrow = lane & 15;  // A row (local j)
  const int nb = lane & 15;    // B column = cp-out / C column

  // Weight matrices in B layout (K padded 16 -> 32 with zeros).
  const float* Wm[4] = {Wm1, Wm2, Wm3, Wm4};
  v16bf Bw[4];
#pragma unroll
  for (int w = 0; w < 4; ++w) {
#pragma unroll
    for (int e = 0; e < 16; ++e) {
      const float v = (e < 8) ? Wm[w][nb * CP + half * 8 + e] : 0.0f;  // B[k][o] = Wm[o][k]
      Bw[w][e] = (bf16_t)v;
    }
  }

  // p_init tile into per-wave LDS (lane: row mrow, cp = half*8 + kk).
  const int j = j0 + mrow;
  const float dx = ref_pos[i * 3] - ref_pos[j * 3];
  const float dy = ref_pos[i * 3 + 1] - ref_pos[j * 3 + 1];
  const float dz = ref_pos[i * 3 + 2] - ref_pos[j * 3 + 2];
  const float inv = 1.0f / (1.0f + dx * dx + dy * dy + dz * dz);
#pragma unroll
  for (int kk = 0; kk < 8; ++kk) {
    const int k = half * 8 + kk;
    sT[wid][mrow][k] = inv * Wd1[k] + Wd2[k] + cu1[i * CP + k] + cu2[j * CP + k];
  }

  v8f cinit;
#pragma unroll
  for (int r = 0; r < 8; ++r) cinit[r] = sT[wid][r + 8 * half][nb];

  v16bf A;
#pragma unroll
  for (int e = 0; e < 16; ++e) {
    const int kA = (e < 8) ? (half * 8 + e) : (8 + half * 8 + e);
    A[e] = (kA < 16) ? (bf16_t)sT[wid][mrow][kA] : (bf16_t)0.0f;
  }

  const v8f z = vzero8();
#pragma unroll
  for (int m = 0; m < 3; ++m) {
    v8f acc = wmma_bf16(A, Bw[m], z);
#pragma unroll
    for (int r = 0; r < 8; ++r) sT[wid][r + 8 * half][nb] = fmaxf(acc[r], 0.0f);
#pragma unroll
    for (int e = 0; e < 16; ++e) {
      const int kA = (e < 8) ? (half * 8 + e) : (8 + half * 8 + e);
      A[e] = (kA < 16) ? (bf16_t)sT[wid][mrow][kA] : (bf16_t)0.0f;
    }
  }
  const v8f pout = wmma_bf16(A, Bw[3], cinit);  // p = p_init + h3 @ Wm4^T
#pragma unroll
  for (int r = 0; r < 8; ++r) {
    const int row = r + 8 * half;
    p_out[(i * NATOM + j0 + row) * CP + nb] = pout[r];
  }
}

// ---------------------------------------------------------------------------
// Projections: qn=LN(q), cn=LN(c); qh/kh/vh.  Pointers pre-offset per block.
// grid=N, block=128.
// ---------------------------------------------------------------------------
__global__ void k_proj(const float* __restrict__ q, const float* __restrict__ c,
                       const float* __restrict__ lnq_g, const float* __restrict__ lnq_b,
                       const float* __restrict__ lnkv_g, const float* __restrict__ lnkv_b,
                       const float* __restrict__ Wq, const float* __restrict__ Wk,
                       const float* __restrict__ Wv,
                       float* __restrict__ qh, float* __restrict__ kh, float* __restrict__ vh) {
  __shared__ float red[128];
  __shared__ float sq[CA];
  __shared__ float sc[CA];
  const int row = blockIdx.x, t = threadIdx.x;
  const float qv = q[row * CA + t];
  const float cv = c[row * CA + t];

  float m = block_reduce_sum128(qv, red) * (1.0f / CA);
  float v = block_reduce_sum128((qv - m) * (qv - m), red) * (1.0f / CA);
  const float qn = (qv - m) * rsqrtf(v + 1e-5f) * lnq_g[t] + lnq_b[t];
  m = block_reduce_sum128(cv, red) * (1.0f / CA);
  v = block_reduce_sum128((cv - m) * (cv - m), red) * (1.0f / CA);
  const float cn = (cv - m) * rsqrtf(v + 1e-5f) * lnkv_g[t] + lnkv_b[t];
  sq[t] = qn;
  sc[t] = cn;
  __syncthreads();

  float aq = 0.0f, ak = 0.0f, av = 0.0f;
  const float* wq = Wq + t * CA;
  const float* wk = Wk + t * CA;
  const float* wv = Wv + t * CA;
  for (int k = 0; k < CA; ++k) {
    aq += wq[k] * sq[k];
    ak += wk[k] * sc[k];
    av += wv[k] * sc[k];
  }
  qh[row * CA + t] = aq;
  kh[row * CA + t] = ak;
  vh[row * CA + t] = av;
}

// ---------------------------------------------------------------------------
// Flash attention partials with pair bias, split-j for parallelism.
// grid = (N/16, JSPLIT), block = 128 (wave w = head w).
// Emits undivided o accumulators + per-row (m, l) for its j-range.
// ---------------------------------------------------------------------------
__global__ void k_attn_part(const float* __restrict__ qh, const float* __restrict__ kh,
                            const float* __restrict__ vh, const float* __restrict__ p,
                            const float* __restrict__ Wb,  // t_Wb + blk*H*CP
                            float* __restrict__ o_part,    // [JSPLIT][N][CA]
                            float* __restrict__ m_part,    // [JSPLIT][NH][N]
                            float* __restrict__ l_part) {  // [JSPLIT][NH][N]
  __shared__ float sWb[NH][CP];
  __shared__ float sB[NH][16][32];
  __shared__ float sS[NH][16][32];
  __shared__ float sAlpha[NH][16];
  const int tid = threadIdx.x;
  const int lane = tid & 31, h = tid >> 5;
  const int half = lane >> 4, mrow = lane & 15, nb = lane & 15;
  const int i0 = blockIdx.x * 16;
  const int js = blockIdx.y;
  const int jbeg = js * (NATOM / JSPLIT);
  const int jend = jbeg + (NATOM / JSPLIT);
  if (tid < NH * CP) sWb[tid >> 4][tid & 15] = Wb[tid];
  const float scale = 0.17677669529663687f;  // 1/sqrt(dh=32)

  // A = q-head tile (16 rows x K=32), loaded once.
  v16bf Aq;
#pragma unroll
  for (int e = 0; e < 16; ++e) {
    const int kA = (e < 8) ? (half * 8 + e) : (8 + half * 8 + e);
    Aq[e] = (bf16_t)qh[(i0 + mrow) * CA + h * DH + kA];
  }

  const v8f z = vzero8();
  v8f oc0 = vzero8(), oc1 = vzero8();
  float mr = -3.0e38f, lr = 0.0f;

  for (int jc = jbeg; jc < jend; jc += 32) {
    __syncthreads();  // previous sB consumers done
    // Cooperative bias staging: 16x32 pair sub-tile of p, all 4 head dots.
#pragma unroll
    for (int pp = 0; pp < 4; ++pp) {
      const int idx = tid + 128 * pp;
      const int ii = idx >> 5, jj = idx & 31;
      const float* pr = p + ((i0 + ii) * NATOM + jc + jj) * CP;
      if (jc + 32 < jend)  // prefetch next chunk's p rows (global_prefetch_b8)
        __builtin_prefetch(pr + 32 * CP, 0, 1);
      float b0 = 0.f, b1 = 0.f, b2 = 0.f, b3 = 0.f;
#pragma unroll
      for (int cp = 0; cp < CP; ++cp) {
        const float pv = pr[cp];
        b0 += pv * sWb[0][cp];
        b1 += pv * sWb[1][cp];
        b2 += pv * sWb[2][cp];
        b3 += pv * sWb[3][cp];
      }
      sB[0][ii][jj] = b0; sB[1][ii][jj] = b1; sB[2][ii][jj] = b2; sB[3][ii][jj] = b3;
    }
    __syncthreads();

    // Scores: two 16x16 tiles (j sub-tiles), K = dh = 32.
#pragma unroll
    for (int sub = 0; sub < 2; ++sub) {
      const int jt = jc + sub * 16;
      v16bf Bk;
#pragma unroll
      for (int e = 0; e < 16; ++e) {
        const int kB = (e < 8) ? (half * 8 + e) : (8 + half * 8 + e);  // B[k][j] = kh[j][k]
        Bk[e] = (bf16_t)kh[(jt + nb) * CA + h * DH + kB];
      }
      const v8f sc = wmma_bf16(Aq, Bk, z);
#pragma unroll
      for (int r = 0; r < 8; ++r) {
        const int rr = r + 8 * half;
        sS[h][rr][sub * 16 + nb] = sc[r] * scale + sB[h][rr][sub * 16 + nb];
      }
    }

    // Online softmax over this 32-wide chunk (lanes 0..15 own rows).
    if (lane < 16) {
      float mc = -3.0e38f;
#pragma unroll
      for (int jj = 0; jj < 32; ++jj) mc = fmaxf(mc, sS[h][lane][jj]);
      const float mn = fmaxf(mr, mc);
      const float alpha = __expf(mr - mn);
      float ps = 0.0f;
#pragma unroll
      for (int jj = 0; jj < 32; ++jj) {
        const float pe = __expf(sS[h][lane][jj] - mn);
        sS[h][lane][jj] = pe;
        ps += pe;
      }
      lr = lr * alpha + ps;
      mr = mn;
      sAlpha[h][lane] = alpha;
    }
    // Rescale running output accumulators (same-wave LDS is in-order).
#pragma unroll
    for (int r = 0; r < 8; ++r) {
      const float al = sAlpha[h][r + 8 * half];
      oc0[r] *= al;
      oc1[r] *= al;
    }
    // P (16x32) x V (32 x dh) as two WMMAs (dh split 16+16).
    v16bf Ap;
#pragma unroll
    for (int e = 0; e < 16; ++e) {
      const int kA = (e < 8) ? (half * 8 + e) : (8 + half * 8 + e);
      Ap[e] = (bf16_t)sS[h][mrow][kA];
    }
    v16bf Bv0, Bv1;
#pragma unroll
    for (int e = 0; e < 16; ++e) {
      const int kB = (e < 8) ? (half * 8 + e) : (8 + half * 8 + e);
      const float* vrow = vh + (jc + kB) * CA + h * DH;
      Bv0[e] = (bf16_t)vrow[nb];
      Bv1[e] = (bf16_t)vrow[16 + nb];
    }
    oc0 = wmma_bf16(Ap, Bv0, oc0);
    oc1 = wmma_bf16(Ap, Bv1, oc1);
  }

  if (lane < 16) {
    m_part[(js * NH + h) * NATOM + i0 + lane] = mr;
    l_part[(js * NH + h) * NATOM + i0 + lane] = lr;
  }
  float* op = o_part + (size_t)js * NATOM * CA;
#pragma unroll
  for (int r = 0; r < 8; ++r) {
    const int row = r + 8 * half;
    op[(i0 + row) * CA + h * DH + nb] = oc0[r];
    op[(i0 + row) * CA + h * DH + 16 + nb] = oc1[r];
  }
}

// ---------------------------------------------------------------------------
// Combine split-j partials: o = sum_js exp(m_js - M) * o_js / L.
// grid=N, block=128.
// ---------------------------------------------------------------------------
__global__ void k_attn_combine(const float* __restrict__ o_part,
                               const float* __restrict__ m_part,
                               const float* __restrict__ l_part,
                               float* __restrict__ obuf) {
  const int row = blockIdx.x, t = threadIdx.x;
  const int h = t >> 5;  // head of this column (DH = 32)
  float mjs[JSPLIT];
  float M = -3.0e38f;
#pragma unroll
  for (int js = 0; js < JSPLIT; ++js) {
    mjs[js] = m_part[(js * NH + h) * NATOM + row];
    M = fmaxf(M, mjs[js]);
  }
  float L = 0.0f, o = 0.0f;
#pragma unroll
  for (int js = 0; js < JSPLIT; ++js) {
    const float w = __expf(mjs[js] - M);
    L += w * l_part[(js * NH + h) * NATOM + row];
    o += w * o_part[(size_t)js * NATOM * CA + row * CA + t];
  }
  obuf[row * CA + t] = o / L;
}

// ---------------------------------------------------------------------------
// Post: q += o@Wo^T; tn = LN(q); q += relu(tn@W1^T)@W2^T.  grid=N, block=128.
// ---------------------------------------------------------------------------
__global__ void k_post(float* __restrict__ qio, const float* __restrict__ obuf,
                       const float* __restrict__ Wo,
                       const float* __restrict__ lnt_g, const float* __restrict__ lnt_b,
                       const float* __restrict__ W1, const float* __restrict__ W2) {
  __shared__ float red[128];
  __shared__ float so[CA];
  __shared__ float st[CA];
  __shared__ float sh[4 * CA];
  const int row = blockIdx.x, t = threadIdx.x;
  so[t] = obuf[row * CA + t];
  __syncthreads();

  float qv = qio[row * CA + t];
  {
    float a = 0.0f;
    const float* wo = Wo + t * CA;
    for (int k = 0; k < CA; ++k) a += wo[k] * so[k];
    qv += a;
  }
  const float m = block_reduce_sum128(qv, red) * (1.0f / CA);
  const float v = block_reduce_sum128((qv - m) * (qv - m), red) * (1.0f / CA);
  st[t] = (qv - m) * rsqrtf(v + 1e-5f) * lnt_g[t] + lnt_b[t];
  __syncthreads();

#pragma unroll
  for (int oo = 0; oo < 4; ++oo) {
    const int o = t + 128 * oo;
    float a = 0.0f;
    const float* w1 = W1 + o * CA;
    for (int k = 0; k < CA; ++k) a += w1[k] * st[k];
    sh[o] = fmaxf(a, 0.0f);
  }
  __syncthreads();

  float a2 = 0.0f;
  const float* w2 = W2 + t * (4 * CA);
  for (int k = 0; k < 4 * CA; ++k) a2 += w2[k] * sh[k];
  qio[row * CA + t] = qv + a2;
}

// ---------------------------------------------------------------------------
// Aggregation: a[t] = mean_{n in token t} (q[n] @ Wagg^T). grid=T, block=384.
// ---------------------------------------------------------------------------
__global__ void k_agg(const float* __restrict__ q, const float* __restrict__ Wagg,
                      float* __restrict__ a_out) {
  __shared__ float sqs[CA];
  const int tok = blockIdx.x;
  const int div = NATOM / TTOK + 1;  // 33
  const int start = tok * div;
  int cnt = NATOM - start;
  if (cnt < 0) cnt = 0;
  if (cnt > div) cnt = div;
  if (threadIdx.x < CA) {
    float s = 0.0f;
    for (int r = 0; r < cnt; ++r) s += q[(start + r) * CA + threadIdx.x];
    sqs[threadIdx.x] = s;
  }
  __syncthreads();
  const int ct = threadIdx.x;  // 0..383
  float acc = 0.0f;
  const float* wa = Wagg + ct * CA;
  for (int k = 0; k < CA; ++k) acc += wa[k] * sqs[k];
  a_out[tok * CT + ct] = acc / (float)(cnt > 0 ? cnt : 1);
}

// ---------------------------------------------------------------------------
extern "C" void kernel_launch(void* const* d_in, const int* in_sizes, int n_in,
                              void* d_out, int out_size, void* d_ws, size_t ws_size,
                              hipStream_t stream) {
  (void)in_sizes; (void)n_in; (void)out_size; (void)ws_size;
  const float* ref_pos = (const float*)d_in[0];
  const float* r_t     = (const float*)d_in[1];
  const float* s_trunk = (const float*)d_in[2];
  // d_in[3] = z_ij (unused by reference)
  const float* W_feat  = (const float*)d_in[4];
  const float* Wd1     = (const float*)d_in[5];
  const float* Wd2     = (const float*)d_in[6];
  const float* ln_ts_g = (const float*)d_in[7];
  const float* ln_ts_b = (const float*)d_in[8];
  const float* Wts     = (const float*)d_in[9];
  const float* Wnp     = (const float*)d_in[10];
  const float* Wpu1    = (const float*)d_in[11];
  const float* Wpu2    = (const float*)d_in[12];
  const float* Wm1     = (const float*)d_in[13];
  const float* Wm2     = (const float*)d_in[14];
  const float* Wm3     = (const float*)d_in[15];
  const float* Wm4     = (const float*)d_in[16];
  const float* t_lnq_g = (const float*)d_in[17];
  const float* t_lnq_b = (const float*)d_in[18];
  const float* t_lnkv_g= (const float*)d_in[19];
  const float* t_lnkv_b= (const float*)d_in[20];
  const float* t_Wq    = (const float*)d_in[21];
  const float* t_Wk    = (const float*)d_in[22];
  const float* t_Wv    = (const float*)d_in[23];
  const float* t_Wo    = (const float*)d_in[24];
  const float* t_Wb    = (const float*)d_in[25];
  const float* t_lnt_g = (const float*)d_in[26];
  const float* t_lnt_b = (const float*)d_in[27];
  const float* t_W1    = (const float*)d_in[28];
  const float* t_W2    = (const float*)d_in[29];
  const float* Wagg    = (const float*)d_in[30];

  float* out = (float*)d_out;
  float* a_out = out;                                   // (T, CT)
  float* q_out = out + TTOK * CT;                       // (N, CA)
  float* c_out = q_out + NATOM * CA;                    // (N, CA)
  float* p_out = c_out + NATOM * CA;                    // (N, N, CP)

  float* ws = (float*)d_ws;
  float* cu1    = ws;
  float* cu2    = cu1 + NATOM * CP;
  float* qh     = cu2 + NATOM * CP;
  float* kh     = qh + NATOM * CA;
  float* vh     = kh + NATOM * CA;
  float* obuf   = vh + NATOM * CA;
  float* o_part = obuf + NATOM * CA;                    // JSPLIT * N * CA
  float* m_part = o_part + (size_t)JSPLIT * NATOM * CA; // JSPLIT * NH * N
  float* l_part = m_part + JSPLIT * NH * NATOM;         // JSPLIT * NH * N

  k_stage1<<<NATOM, 128, 0, stream>>>(ref_pos, r_t, s_trunk, W_feat, ln_ts_g, ln_ts_b,
                                      Wts, Wnp, Wpu1, Wpu2, q_out, c_out, cu1, cu2);

  k_pair<<<(NATOM * NTILE) / 4, 128, 0, stream>>>(ref_pos, Wd1, Wd2,
                                                  Wm1, Wm2, Wm3, Wm4,
                                                  cu1, cu2, p_out);

  for (int blk = 0; blk < NB; ++blk) {
    k_proj<<<NATOM, 128, 0, stream>>>(q_out, c_out,
        t_lnq_g + blk * CA, t_lnq_b + blk * CA,
        t_lnkv_g + blk * CA, t_lnkv_b + blk * CA,
        t_Wq + blk * CA * CA, t_Wk + blk * CA * CA, t_Wv + blk * CA * CA,
        qh, kh, vh);
    k_attn_part<<<dim3(NTILE, JSPLIT), 128, 0, stream>>>(qh, kh, vh, p_out,
        t_Wb + blk * NH * CP, o_part, m_part, l_part);
    k_attn_combine<<<NATOM, 128, 0, stream>>>(o_part, m_part, l_part, obuf);
    k_post<<<NATOM, 128, 0, stream>>>(q_out, obuf, t_Wo + blk * CA * CA,
        t_lnt_g + blk * CA, t_lnt_b + blk * CA,
        t_W1 + blk * 4 * CA * CA, t_W2 + blk * CA * 4 * CA);
  }

  k_agg<<<TTOK, 384, 0, stream>>>(q_out, Wagg, a_out);
}
